// NeRFNetworkBasis_77111842832972
// MI455X (gfx1250) — compile-verified, matrix operand order checked by reading, not run on hardware
//
#include <hip/hip_runtime.h>

// ---------------------------------------------------------------------------
// NeRF density-grid update (instant-ngp style), MI455X / gfx1250.
// Memory-bound: ~1.05 GB of traffic -> ~45us floor at 23.3 TB/s.
//   k_update : fused morton-gather + decay/max merge + per-block partial sum
//              (block reduction done with one v_wmma_f32_16x16x32_f16)
//   k_thresh : fold block partials -> thresh = min(mean, 0.01)
//   k_pack   : threshold + packbits (8 cells/byte), NT stores
// Output layout (float*): [new_grid : 67,108,864 f32][bitfield : 8,388,608
// values 0..255 stored as float (integer-valued)].
// ---------------------------------------------------------------------------

typedef __attribute__((ext_vector_type(16))) _Float16 v16h;
typedef __attribute__((ext_vector_type(8)))  float    v8f;
typedef __attribute__((ext_vector_type(4)))  float    v4f;
typedef __attribute__((ext_vector_type(2)))  float    v2f;

#define G3_LOG2 21
#define G3      (1u << G3_LOG2)          // 128^3
#define NTOT    ((size_t)32 << G3_LOG2)  // 67,108,864 elements
#define NGROUPS (1u << 23)               // NTOT / 8
#define NQUADS  (1u << 21)               // bitfield bytes / 4
#define BLOCKS  8192
#define THREADS 256

// inverse of the reference's 3-way bit expansion (morton decode, 7-bit coords)
__device__ __forceinline__ unsigned compact3(unsigned v) {
  v &= 0x49249249u;
  v = (v | (v >> 2))  & 0xC30C30C3u;
  v = (v | (v >> 4))  & 0x0F00F00Fu;
  v = (v | (v >> 8))  & 0xFF0000FFu;
  v = (v | (v >> 16)) & 0x0000FFFFu;
  return v & 0x7Fu;
}

__global__ __launch_bounds__(THREADS)
void k_update(const float* __restrict__ dg, const float* __restrict__ sig,
              float* __restrict__ out, float* __restrict__ partial)
{
  const unsigned tid    = blockIdx.x * THREADS + threadIdx.x;
  const unsigned stride = gridDim.x * THREADS;
  float acc = 0.f;

  // Each group = 8 consecutive morton codes = a 2x2x2 spatial block.
  for (unsigned g = tid; g < NGROUPS; g += stride) {
    const unsigned e = g << 3;               // flat element index (< 2^26)
    const unsigned t = e >> G3_LOG2;
    const unsigned j = e & (G3 - 1u);        // morton index, multiple of 8
    const float* dgt = dg  + ((size_t)t << G3_LOG2);
    const float* sgt = sig + ((size_t)t << G3_LOG2);
    float*       ot  = out + ((size_t)t << G3_LOG2);

    // coalesced b128 reads of density_grid (read-once -> NT)
    v4f d0 = __builtin_nontemporal_load((const v4f*)(dgt + j));
    v4f d1 = __builtin_nontemporal_load((const v4f*)(dgt + j + 4));

    // sigma gather: base of the 2x2x2 block, then 4 contiguous float2 pairs
    const unsigned xb = compact3(j);
    const unsigned yb = compact3(j >> 1);
    const unsigned zb = compact3(j >> 2);
    const unsigned sb = xb * 16384u + yb * 128u + zb;

    v2f p0 = __builtin_nontemporal_load((const v2f*)(sgt + sb));          // c0,c4
    v2f p1 = __builtin_nontemporal_load((const v2f*)(sgt + sb + 128));    // c2,c6
    v2f p2 = __builtin_nontemporal_load((const v2f*)(sgt + sb + 16384));  // c1,c5
    v2f p3 = __builtin_nontemporal_load((const v2f*)(sgt + sb + 16512));  // c3,c7

    const float dv[8] = { d0[0], d0[1], d0[2], d0[3], d1[0], d1[1], d1[2], d1[3] };
    const float sv[8] = { p0[0], p2[0], p1[0], p3[0], p0[1], p2[1], p1[1], p3[1] };

    float ov[8];
#pragma unroll
    for (int k = 0; k < 8; ++k) {
      const float d = dv[k];
      const float m = fmaxf(d * 0.95f, sv[k]);
      const float o = (d >= 0.f) ? m : d;    // sigmas are always >= 0
      ov[k] = o;
      acc += fmaxf(o, 0.f);                  // clip(new,0) for the mean
    }
    v4f o0 = { ov[0], ov[1], ov[2], ov[3] };
    v4f o1 = { ov[4], ov[5], ov[6], ov[7] };
    *(v4f*)(ot + j)     = o0;                // RT: re-read by k_pack via L2
    *(v4f*)(ot + j + 4) = o1;
  }

  // ---- block reduction of 256 thread partials with ONE WMMA -------------
  // A (16x32 f16) holds the 256 partials (+256 zeros), B = all ones, C = 0.
  // With B all-ones, sum(D) = 16 * sum(A) independent of the VGPR layout.
  __shared__ float smem[THREADS];
  smem[threadIdx.x] = acc;
  __syncthreads();

  const unsigned lane = threadIdx.x & 31u;
  v16h a;
#pragma unroll
  for (int i = 0; i < 8; ++i)  a[i] = (_Float16)smem[lane * 8u + i];
#pragma unroll
  for (int i = 8; i < 16; ++i) a[i] = (_Float16)0.f;
  v16h b;
#pragma unroll
  for (int i = 0; i < 16; ++i) b[i] = (_Float16)1.f;
  v8f c = {};
  c = __builtin_amdgcn_wmma_f32_16x16x32_f16(false, a, false, b,
                                             (short)0, c, false, false);
  float s = c[0] + c[1] + c[2] + c[3] + c[4] + c[5] + c[6] + c[7];
#pragma unroll
  for (int off = 16; off >= 1; off >>= 1) s += __shfl_xor(s, off, 32);
  s *= (1.0f / 16.0f);                       // undo the 16x column replication

  if (threadIdx.x == 0) partial[blockIdx.x] = s;
}

__global__ __launch_bounds__(THREADS)
void k_thresh(const float* __restrict__ partial, float* __restrict__ thresh)
{
  __shared__ float sm[THREADS];
  float a = 0.f;
  for (unsigned i = threadIdx.x; i < (unsigned)BLOCKS; i += THREADS)
    a += partial[i];
  sm[threadIdx.x] = a;
  __syncthreads();
#pragma unroll
  for (unsigned off = THREADS / 2; off > 0; off >>= 1) {
    if (threadIdx.x < off) sm[threadIdx.x] += sm[threadIdx.x + off];
    __syncthreads();
  }
  if (threadIdx.x == 0) {
    const float mean = sm[0] * (1.0f / 67108864.0f);
    thresh[0] = fminf(mean, 0.01f);
  }
}

__global__ __launch_bounds__(THREADS)
void k_pack(const float* __restrict__ ng, const float* __restrict__ thresh,
            float* __restrict__ bits)
{
  const float th = thresh[0];
  const unsigned tid    = blockIdx.x * THREADS + threadIdx.x;
  const unsigned stride = gridDim.x * THREADS;

  for (unsigned q = tid; q < NQUADS; q += stride) {   // 4 output bytes / thread
    const size_t base = (size_t)q * 32;
    v4f bo;
#pragma unroll
    for (int byte = 0; byte < 4; ++byte) {
      v4f a0 = *(const v4f*)(ng + base + byte * 8);     // hot in L2 from k_update
      v4f a1 = *(const v4f*)(ng + base + byte * 8 + 4);
      int v = 0;
#pragma unroll
      for (int k = 0; k < 4; ++k) v |= (a0[k] > th) ? (1 << k) : 0;
#pragma unroll
      for (int k = 0; k < 4; ++k) v |= (a1[k] > th) ? (1 << (k + 4)) : 0;
      bo[byte] = (float)v;                              // integer-valued float
    }
    __builtin_nontemporal_store(bo, (v4f*)(bits + (size_t)q * 4));
  }
}

extern "C" void kernel_launch(void* const* d_in, const int* in_sizes, int n_in,
                              void* d_out, int out_size, void* d_ws, size_t ws_size,
                              hipStream_t stream)
{
  (void)in_sizes; (void)n_in; (void)out_size; (void)ws_size;

  const float* dg  = (const float*)d_in[0];   // density_grid [32,1,128^3]
  const float* sig = (const float*)d_in[1];   // sigmas       [32,1,128^3]

  float* out_grid = (float*)d_out;            // new_grid, 67,108,864 f32
  float* out_bits = (float*)d_out + NTOT;     // bitfield, 8,388,608 values
  float* partial  = (float*)d_ws;             // BLOCKS block partial sums
  float* thresh   = partial + BLOCKS;         // 1 float

  k_update<<<BLOCKS, THREADS, 0, stream>>>(dg, sig, out_grid, partial);
  k_thresh<<<1,      THREADS, 0, stream>>>(partial, thresh);
  k_pack  <<<BLOCKS, THREADS, 0, stream>>>(out_grid, thresh, out_bits);
}
